// GptOssAttention_738734375613
// MI455X (gfx1250) — matrix operand order repaired and checked by reading, not compiled, not run on hardware
//
#include <hip/hip_runtime.h>
#include <hip/hip_bf16.h>
#include <math.h>

typedef __attribute__((ext_vector_type(16))) __bf16          v16bf;
typedef __attribute__((ext_vector_type(8)))  float           v8f;
typedef __attribute__((ext_vector_type(4)))  float           f4;
typedef __attribute__((ext_vector_type(8)))  unsigned short  us8;
typedef __attribute__((ext_vector_type(4)))  unsigned short  us4;

union FragBF { us8 h[2]; v16bf v; };
union U4x32  { unsigned int w[4]; us8 h; };
union U2x32  { unsigned int w[2]; us4 h; };

__device__ __forceinline__ unsigned short f2bf(float f) {
    unsigned int u = __float_as_uint(f);
    u += 0x7FFFu + ((u >> 16) & 1u);      // round-to-nearest-even
    return (unsigned short)(u >> 16);
}

// packed f32x2 -> bf16x2 (V_CVT_PK_BF16_F32 when available)
#if __has_builtin(__builtin_amdgcn_cvt_pk_bf16_f32)
__device__ __forceinline__ unsigned int pk_bf16(float a, float b) {
    auto r = __builtin_amdgcn_cvt_pk_bf16_f32(a, b);
    return __builtin_bit_cast(unsigned int, r);
}
#else
__device__ __forceinline__ unsigned int pk_bf16(float a, float b) {
    return (unsigned int)f2bf(a) | ((unsigned int)f2bf(b) << 16);
}
#endif

// XOR-lane reduction step via ds_swizzle_b32 (group-of-32: pat=(xor<<10)|0x1F)
template <int PAT>
__device__ __forceinline__ float swzx(float x) {
#if __has_builtin(__builtin_amdgcn_ds_swizzle)
    return __int_as_float(__builtin_amdgcn_ds_swizzle(__float_as_int(x), PAT));
#else
    return __shfl_xor(x, (PAT >> 10) & 31);
#endif
}
__device__ __forceinline__ float redmax16(float v) {
    v = fmaxf(v, swzx<0x041F>(v));
    v = fmaxf(v, swzx<0x081F>(v));
    v = fmaxf(v, swzx<0x101F>(v));
    v = fmaxf(v, swzx<0x201F>(v));
    return v;
}
__device__ __forceinline__ float redsum16(float v) {
    v += swzx<0x041F>(v);
    v += swzx<0x081F>(v);
    v += swzx<0x101F>(v);
    v += swzx<0x201F>(v);
    return v;
}

__device__ __forceinline__ v8f v8f_zero() {
    v8f z;
    for (int i = 0; i < 8; ++i) z[i] = 0.f;
    return z;
}
__device__ __forceinline__ us8 us8_zero() {
    us8 z;
    for (int i = 0; i < 8; ++i) z[i] = 0;
    return z;
}

// ---------------------------------------------------------------------------
// Tiled bf16 WMMA GEMM:  C[M,N] = A[M,K] * B[K,N] + bias   (fp32 out)
// Block tile 128x128, K-step 32, 8 waves (256 thr), wave tile 32x64.
// ---------------------------------------------------------------------------
#define TM 128
#define TN 128
#define TK 32
#define APITCH (TK + 8)

template <bool A_BF16>
__global__ __launch_bounds__(256)
void gemm_bf16_wmma(const void* __restrict__ Aptr,
                    const float* __restrict__ B,
                    const float* __restrict__ bias,
                    float* __restrict__ C,
                    int N, int K)
{
    __shared__ unsigned short As[TM][APITCH];
    __shared__ unsigned short Bs[TN][APITCH];

    const int m0   = blockIdx.x * TM;
    const int n0   = blockIdx.y * TN;
    const int tid  = threadIdx.x;
    const int lane = tid & 31;
    const int w    = tid >> 5;
    const int wm   = (w & 3) * 32;     // 4 waves along M
    const int wn   = (w >> 2) * 64;    // 2 waves along N
    const int half = lane >> 4;
    const int l16  = lane & 15;

    v8f acc[2][4];
    for (int i = 0; i < 2; ++i)
        for (int j = 0; j < 4; ++j) acc[i][j] = v8f_zero();

    const float*          Af = (const float*)Aptr;
    const unsigned short* Ab = (const unsigned short*)Aptr;

    for (int k0 = 0; k0 < K; k0 += TK) {
        // ---- stage A tile (128 x 32) ----
        if (A_BF16) {
            #pragma unroll
            for (int i = tid; i < TM * TK / 8; i += 256) {
                int r = i >> 2;
                int c = (i & 3) * 8;
                us8 hv = *(const us8*)(Ab + (size_t)(m0 + r) * K + k0 + c);
                *(us8*)&As[r][c] = hv;
            }
        } else {
            #pragma unroll
            for (int i = tid; i < TM * TK / 4; i += 256) {
                int r = i >> 3;
                int c = (i & 7) * 4;
                f4 fv = *(const f4*)(Af + (size_t)(m0 + r) * K + k0 + c);
                U2x32 u;
                u.w[0] = pk_bf16(fv[0], fv[1]);
                u.w[1] = pk_bf16(fv[2], fv[3]);
                *(us4*)&As[r][c] = u.h;
            }
        }
        // ---- stage B tile (32 x 128) transposed -> Bs[n][k] ----
        #pragma unroll
        for (int i = tid; i < 512; i += 256) {
            int n  = i & 127;
            int kg = (i >> 7) * 8;            // 0,8,16,24
            int gn = n0 + n;
            U4x32 u;
            if (gn < N) {
                #pragma unroll
                for (int q = 0; q < 4; ++q)
                    u.w[q] = pk_bf16(B[(size_t)(k0 + kg + 2 * q) * N + gn],
                                     B[(size_t)(k0 + kg + 2 * q + 1) * N + gn]);
            } else {
                u.h = us8_zero();
            }
            *(us8*)&Bs[n][kg] = u.h;
        }
        // prefetch next K tile (global_prefetch_b8)
        if (k0 + TK < K) {
            __builtin_prefetch(A_BF16
                ? (const void*)(Ab + (size_t)(m0 + (tid >> 1)) * K + k0 + TK)
                : (const void*)(Af + (size_t)(m0 + (tid >> 1)) * K + k0 + TK), 0, 1);
            __builtin_prefetch((const void*)(B + (size_t)(k0 + TK + (tid >> 4)) * N + n0), 0, 1);
        }
        __syncthreads();

        FragBF a[2];
        #pragma unroll
        for (int mt = 0; mt < 2; ++mt) {
            int r = wm + mt * 16 + l16;
            a[mt].h[0] = *(const us8*)&As[r][half * 8];
            a[mt].h[1] = *(const us8*)&As[r][16 + half * 8];
        }
        #pragma unroll
        for (int nt = 0; nt < 4; ++nt) {
            FragBF b;
            int cn = wn + nt * 16 + l16;
            b.h[0] = *(const us8*)&Bs[cn][half * 8];
            b.h[1] = *(const us8*)&Bs[cn][16 + half * 8];
            #pragma unroll
            for (int mt = 0; mt < 2; ++mt) {
                acc[mt][nt] = __builtin_amdgcn_wmma_f32_16x16x32_bf16(
                    false, a[mt].v, false, b.v, (short)0, acc[mt][nt],
                    false, false);
            }
        }
        __syncthreads();
    }

    // epilogue: bias add + fp32 store
    #pragma unroll
    for (int mt = 0; mt < 2; ++mt)
        #pragma unroll
        for (int nt = 0; nt < 4; ++nt) {
            int n = n0 + wn + nt * 16 + l16;
            if (n < N) {
                float bv = bias[n];
                #pragma unroll
                for (int j = 0; j < 8; ++j) {
                    int m = m0 + wm + mt * 16 + half * 8 + j;
                    C[(size_t)m * N + n] = acc[mt][nt][j] + bv;
                }
            }
        }
}

// ---------------------------------------------------------------------------
// YaRN RoPE: qkv fp32 (S x 5120) -> q bf16 (S,64,64)*D^-0.5, k bf16 (S,8,64),
// v bf16 (S,8,64). One block per token; packed bf16x2 stores.
// ---------------------------------------------------------------------------
__global__ __launch_bounds__(256)
void rope_yarn_kernel(const float* __restrict__ qkv,
                      const int* __restrict__ pos_ids,
                      unsigned short* __restrict__ qb,
                      unsigned short* __restrict__ kb,
                      unsigned short* __restrict__ vb)
{
    __shared__ float cs[32], sn[32];
    const int t   = blockIdx.x;
    const int tid = threadIdx.x;

    if (tid < 32) {
        const float lnBase = logf(150000.f);
        const float twoPi  = 6.28318530717958647692f;
        float pf     = expf(lnBase * (float)tid / 32.f);
        float extrap = 1.f / pf;
        float interp = 1.f / (32.f * pf);
        float cLow   = 64.f * logf(4096.f / (32.f * twoPi)) / (2.f * lnBase);
        float cHigh  = 64.f * logf(4096.f / (1.f * twoPi)) / (2.f * lnBase);
        float low    = fmaxf(floorf(cLow), 0.f);
        float high   = fminf(ceilf(cHigh), 63.f);
        float ramp   = ((float)tid - low) / fmaxf(high - low, 0.001f);
        ramp         = fminf(fmaxf(ramp, 0.f), 1.f);
        float extM   = 1.f - ramp;
        float invF   = interp * (1.f - extM) + extrap * extM;
        float mscale = 0.1f * logf(32.f) + 1.f;
        float fr     = (float)pos_ids[t] * invF;
        cs[tid] = cosf(fr) * mscale;
        sn[tid] = sinf(fr) * mscale;
    }
    __syncthreads();

    const float* row = qkv + (size_t)t * 5120;
    const float qscale = 0.125f;   // D^-0.5, D=64

    #pragma unroll
    for (int i = tid; i < 64 * 16; i += 256) {      // Q: 64 heads, dim pairs
        int h = i >> 4, jp = (i & 15) * 2;
        float x1a = row[h * 64 + jp],      x1b = row[h * 64 + jp + 1];
        float x2a = row[h * 64 + 32 + jp], x2b = row[h * 64 + 32 + jp + 1];
        float ca = cs[jp], cb = cs[jp + 1], sa = sn[jp], sb = sn[jp + 1];
        size_t base = ((size_t)t * 64 + h) * 64;
        *(unsigned int*)&qb[base + jp] =
            pk_bf16((x1a * ca - x2a * sa) * qscale, (x1b * cb - x2b * sb) * qscale);
        *(unsigned int*)&qb[base + 32 + jp] =
            pk_bf16((x2a * ca + x1a * sa) * qscale, (x2b * cb + x1b * sb) * qscale);
    }
    for (int i = tid; i < 8 * 16; i += 256) {       // K: 8 kv heads, dim pairs
        int h = i >> 4, jp = (i & 15) * 2;
        float x1a = row[4096 + h * 64 + jp],      x1b = row[4096 + h * 64 + jp + 1];
        float x2a = row[4096 + h * 64 + 32 + jp], x2b = row[4096 + h * 64 + 32 + jp + 1];
        float ca = cs[jp], cb = cs[jp + 1], sa = sn[jp], sb = sn[jp + 1];
        size_t base = ((size_t)t * 8 + h) * 64;
        *(unsigned int*)&kb[base + jp] =
            pk_bf16(x1a * ca - x2a * sa, x1b * cb - x2b * sb);
        *(unsigned int*)&kb[base + 32 + jp] =
            pk_bf16(x2a * ca + x1a * sa, x2b * cb + x1b * sb);
    }
    {                                               // V: 512 values = 256 pairs
        int i = tid;
        *(unsigned int*)&vb[(size_t)t * 512 + 2 * i] =
            pk_bf16(row[4608 + 2 * i], row[4608 + 2 * i + 1]);
    }
}

// ---------------------------------------------------------------------------
// Blocked sliding-window attention with sinks.
// Wave w owns query rows 16w..16w+15; valid keys span m in [16w+1, 16w+143],
// so only 10 score tiles (t0 = w&~1) and 5 PV chunks are computed.
// ---------------------------------------------------------------------------
#define KPITCH 72
#define VPITCH 264
#define PPITCH 264

__global__ __launch_bounds__(256)
void attn_swa_kernel(const unsigned short* __restrict__ qb,
                     const unsigned short* __restrict__ kb,
                     const unsigned short* __restrict__ vb,
                     const float* __restrict__ sinks,
                     unsigned short* __restrict__ attn)
{
    extern __shared__ unsigned short lds[];
    unsigned short* ks = lds;                          // 256 x KPITCH
    unsigned short* vs = ks + 256 * KPITCH;            // 64  x VPITCH
    unsigned short* ps = vs + 64 * VPITCH;             // 128 x PPITCH

    const int nblk = blockIdx.x;     // 0..31
    const int head = blockIdx.y;     // 0..63
    const int kvh  = head >> 3;
    const int tid  = threadIdx.x;
    const int lane = tid & 31;
    const int w    = tid >> 5;
    const int half = lane >> 4;
    const int l16  = lane & 15;

    // stage K rows [m][d] (b128) and V transposed [d][m]; zero-pad block -1
    #pragma unroll
    for (int i = tid; i < 2048; i += 256) {
        int m  = i >> 3;             // 0..255
        int dg = (i & 7) * 8;        // 0,8,...,56
        int tok = (nblk - 1) * 128 + m;
        us8 kvv = us8_zero();
        us8 vvv = us8_zero();
        if (tok >= 0) {
            size_t base = ((size_t)tok * 8 + kvh) * 64 + dg;
            kvv = *(const us8*)(kb + base);
            vvv = *(const us8*)(vb + base);
        }
        *(us8*)&ks[m * KPITCH + dg] = kvv;
        #pragma unroll
        for (int q = 0; q < 8; ++q) vs[(dg + q) * VPITCH + m] = vvv[q];
    }
    __syncthreads();

    // Q fragments straight from global (this wave's 16 rows)
    FragBF qf[2];
    {
        int tok = nblk * 128 + 16 * w + l16;
        const unsigned short* qrow = qb + ((size_t)tok * 64 + head) * 64;
        #pragma unroll
        for (int c = 0; c < 2; ++c) {
            qf[c].h[0] = *(const us8*)(qrow + c * 32 + half * 8);
            qf[c].h[1] = *(const us8*)(qrow + c * 32 + 16 + half * 8);
        }
    }

    // scores: only the 10 tiles intersecting the sliding window band
    const int t0    = w & ~1;            // even tile base: tiles t0..t0+9
    const int mbase = t0 * 16 + l16;     // this lane's m for local tile 0
    v8f sc[10];
    #pragma unroll
    for (int t = 0; t < 10; ++t) sc[t] = v8f_zero();
    #pragma unroll
    for (int c = 0; c < 2; ++c) {
        #pragma unroll
        for (int t = 0; t < 10; ++t) {
            FragBF bf_;
            const unsigned short* kr =
                ks + ((t0 + t) * 16 + l16) * KPITCH + c * 32 + half * 8;
            bf_.h[0] = *(const us8*)kr;
            bf_.h[1] = *(const us8*)(kr + 16);
            sc[t] = __builtin_amdgcn_wmma_f32_16x16x32_bf16(
                false, qf[c].v, false, bf_.v, (short)0, sc[t], false, false);
        }
    }

    // band mask applied once, in place (single unsigned range test per elem)
    const int minM = (nblk == 0) ? 128 : 0;
    #pragma unroll
    for (int j = 0; j < 8; ++j) {
        int qrow = 16 * w + 8 * half + j;
        int lo = qrow + 1; if (lo < minM) lo = minM;
        unsigned span = (unsigned)(qrow + 128 - lo);
        #pragma unroll
        for (int t = 0; t < 10; ++t) {
            if ((unsigned)(mbase + t * 16 - lo) > span) sc[t][j] = -1e30f;
        }
    }

    // softmax with sink (ds_swizzle XOR reductions across the 16-lane group)
    const float sinkv = sinks[head];
    float rs[8];
    #pragma unroll
    for (int j = 0; j < 8; ++j) {
        float mx = -1e30f;
        #pragma unroll
        for (int t = 0; t < 10; ++t) mx = fmaxf(mx, sc[t][j]);
        mx = fmaxf(redmax16(mx), sinkv);
        float sum = 0.f;
        const int prow = (w * 16 + 8 * half + j) * PPITCH + mbase;
        #pragma unroll
        for (int t = 0; t < 10; ++t) {
            float p = __expf(sc[t][j] - mx);     // masked -> exactly 0
            sum += p;
            ps[prow + t * 16] = f2bf(p);
        }
        rs[j] = redsum16(sum) + __expf(sinkv - mx);
    }
    __syncthreads();

    // P @ V : out 16x64, K = m restricted to the 5 live chunks of 32
    const int mc0 = t0 >> 1;
    v8f o[4];
    #pragma unroll
    for (int nt = 0; nt < 4; ++nt) o[nt] = v8f_zero();
    #pragma unroll
    for (int cc = 0; cc < 5; ++cc) {
        const int mo = (mc0 + cc) * 32;
        FragBF pa;
        const unsigned short* pr = ps + (w * 16 + l16) * PPITCH + mo;
        pa.h[0] = *(const us8*)(pr + half * 8);
        pa.h[1] = *(const us8*)(pr + 16 + half * 8);
        #pragma unroll
        for (int nt = 0; nt < 4; ++nt) {
            FragBF vf;
            const unsigned short* vr =
                vs + (nt * 16 + l16) * VPITCH + mo + half * 8;
            vf.h[0] = *(const us8*)vr;
            vf.h[1] = *(const us8*)(vr + 16);
            o[nt] = __builtin_amdgcn_wmma_f32_16x16x32_bf16(
                false, pa.v, false, vf.v, (short)0, o[nt], false, false);
        }
    }

    // normalize and store bf16 attn[token][head*64 + d]
    #pragma unroll
    for (int nt = 0; nt < 4; ++nt) {
        int d = nt * 16 + l16;
        #pragma unroll
        for (int j = 0; j < 8; ++j) {
            int tok = nblk * 128 + 16 * w + 8 * half + j;
            attn[(size_t)tok * 4096 + head * 64 + d] = f2bf(o[nt][j] / rs[j]);
        }
    }
}

// ---------------------------------------------------------------------------
extern "C" void kernel_launch(void* const* d_in, const int* in_sizes, int n_in,
                              void* d_out, int out_size, void* d_ws, size_t ws_size,
                              hipStream_t stream)
{
    (void)in_sizes; (void)n_in; (void)out_size; (void)ws_size;

    const float* hidden = (const float*)d_in[0];
    const int*   pos    = (const int*)d_in[1];
    const float* w_qkv  = (const float*)d_in[2];
    const float* b_qkv  = (const float*)d_in[3];
    const float* w_o    = (const float*)d_in[4];
    const float* b_o    = (const float*)d_in[5];
    const float* sinks  = (const float*)d_in[6];

    char* ws = (char*)d_ws;
    float*          qkv  = (float*)ws;                              // 4096*5120 f32
    unsigned short* qb   = (unsigned short*)(ws + 4096UL * 5120 * 4);
    unsigned short* kb   = qb + 4096UL * 64 * 64;
    unsigned short* vb   = kb + 4096UL * 8 * 64;
    unsigned short* attn = vb + 4096UL * 8 * 64;                    // 4096*4096 bf16

    // 1) QKV projection: (4096x2880) x (2880x5120) + b_qkv
    dim3 g1(4096 / TM, 5120 / TN);
    gemm_bf16_wmma<false><<<g1, 256, 0, stream>>>(hidden, w_qkv, b_qkv, qkv,
                                                  5120, 2880);

    // 2) YaRN RoPE + bf16 pack
    rope_yarn_kernel<<<4096, 256, 0, stream>>>(qkv, pos, qb, kb, vb);

    // 3) sliding-window attention with sinks
    size_t attn_lds = (size_t)(256 * KPITCH + 64 * VPITCH + 128 * PPITCH) *
                      sizeof(unsigned short);
    dim3 g2(32, 64);
    attn_swa_kernel<<<g2, 256, attn_lds, stream>>>(qb, kb, vb, sinks, attn);

    // 4) output projection: (4096x4096 bf16) x (4096x2880) + b_o
    dim3 g3(4096 / TM, (2880 + TN - 1) / TN);
    gemm_bf16_wmma<true><<<g3, 256, 0, stream>>>(attn, w_o, b_o, (float*)d_out,
                                                 2880, 4096);
}
